// FADEv4_7026566496861
// MI455X (gfx1250) — compile-verified, matrix operand order checked by reading, not run on hardware
//
#include <hip/hip_runtime.h>
#include <hip/hip_bf16.h>
#include <math.h>

typedef _Float16 v16h __attribute__((ext_vector_type(16)));
typedef _Float16 h8   __attribute__((ext_vector_type(8)));
typedef _Float16 h4   __attribute__((ext_vector_type(4)));
typedef float    v8f  __attribute__((ext_vector_type(8)));

#define D_DIM 768
#define N_B   16
#define TQ    1369
#define TS    5476
#define NSHOT 4

__device__ __forceinline__ float wave_sum32(float v) {
    #pragma unroll
    for (int off = 16; off >= 1; off >>= 1) v += __shfl_xor(v, off, 32);
    return v;
}

// One wave (32 lanes) per row: L2-normalize row of length 768, emit f16 row
// and the scalar dot(normalized_row, w_seg) for the linear head.
__global__ void norm_rows_kernel(const float* __restrict__ x,
                                 const float* __restrict__ wseg,
                                 _Float16* __restrict__ xh,
                                 float* __restrict__ hout,
                                 int rows) {
    const int wave = threadIdx.x >> 5;
    const int lane = threadIdx.x & 31;
    const long row = (long)blockIdx.x * 8 + wave;   // blockDim.x == 256 -> 8 waves
    if (row >= rows) return;

    const float4* xp = (const float4*)(x + row * (long)D_DIM);
    const float4* wp = (const float4*)wseg;

    float v[24];
    float ss = 0.f;
    #pragma unroll
    for (int i = 0; i < 6; ++i) {
        float4 t = xp[lane + 32 * i];
        v[4*i+0] = t.x; v[4*i+1] = t.y; v[4*i+2] = t.z; v[4*i+3] = t.w;
        ss += t.x*t.x + t.y*t.y + t.z*t.z + t.w*t.w;
    }
    ss = wave_sum32(ss);
    const float inv = 1.0f / fmaxf(sqrtf(ss), 1e-12f);

    _Float16* op = xh + row * (long)D_DIM;
    float hd = 0.f;
    #pragma unroll
    for (int i = 0; i < 6; ++i) {
        float a = v[4*i+0]*inv, b = v[4*i+1]*inv, c = v[4*i+2]*inv, d = v[4*i+3]*inv;
        float4 w = wp[lane + 32 * i];
        hd += a*w.x + b*w.y + c*w.z + d*w.w;
        h4 o = { (_Float16)a, (_Float16)b, (_Float16)c, (_Float16)d };
        ((h4*)op)[lane + 32 * i] = o;
    }
    hd = wave_sum32(hd);
    if (lane == 0) hout[row] = hd;
}

// One wave per batch element n: mean over shots -> L2 norm -> dot with W seg 3.
__global__ void cls_head_kernel(const float* __restrict__ xcls,
                                const float* __restrict__ wseg,
                                float* __restrict__ hcls) {
    const int n = blockIdx.x;
    const int lane = threadIdx.x & 31;
    const float4* base = (const float4*)(xcls + (long)n * NSHOT * D_DIM);
    const float4* wp = (const float4*)wseg;

    float m[24];
    float ss = 0.f;
    #pragma unroll
    for (int i = 0; i < 6; ++i) {
        int idx = lane + 32 * i;
        float4 a0 = base[idx];
        float4 a1 = base[idx + (D_DIM/4)];
        float4 a2 = base[idx + 2*(D_DIM/4)];
        float4 a3 = base[idx + 3*(D_DIM/4)];
        float mx = 0.25f*(a0.x+a1.x+a2.x+a3.x);
        float my = 0.25f*(a0.y+a1.y+a2.y+a3.y);
        float mz = 0.25f*(a0.z+a1.z+a2.z+a3.z);
        float mw = 0.25f*(a0.w+a1.w+a2.w+a3.w);
        m[4*i+0]=mx; m[4*i+1]=my; m[4*i+2]=mz; m[4*i+3]=mw;
        ss += mx*mx + my*my + mz*mz + mw*mw;
    }
    ss = wave_sum32(ss);
    const float inv = 1.0f / fmaxf(sqrtf(ss), 1e-12f);

    float hd = 0.f;
    #pragma unroll
    for (int i = 0; i < 6; ++i) {
        float4 w = wp[lane + 32 * i];
        hd += m[4*i+0]*inv*w.x + m[4*i+1]*inv*w.y + m[4*i+2]*inv*w.z + m[4*i+3]*inv*w.w;
    }
    hd = wave_sum32(hd);
    if (lane == 0) hcls[n] = hd;
}

// 4 waves/block; each wave owns TWO 16-row query tiles (32 rows) and each
// iteration covers TWO 16-col support tiles staged in 48 KB of LDS -> a 2x2
// WMMA register tile, so every A fragment and every B fragment feeds two
// v_wmma_f32_16x16x32_f16 ops (1 VMEM + 1 LDS load per WMMA). Running
// max-similarity / argmax is fused in registers; head + sigmoid finalized here.
__global__ void __launch_bounds__(128)
sim_argmax_kernel(const _Float16* __restrict__ qh,
                  const _Float16* __restrict__ sh,
                  const float* __restrict__ hq,
                  const float* __restrict__ hs,
                  const float* __restrict__ hcls,
                  const float* __restrict__ bhead,
                  float* __restrict__ out0,
                  float* __restrict__ out1) {
    __shared__ _Float16 Bbuf[32 * D_DIM];   // 48 KB: 32 support rows

    const int n    = blockIdx.y;
    const int wave = threadIdx.x >> 5;
    const int lane = threadIdx.x & 31;
    const int m_base = blockIdx.x * 128 + wave * 32;   // may exceed TQ: clamp, no exit
    const int half = lane >> 4;        // K sub-chunk / C-row half selector
    const int l16  = lane & 15;

    const int arow0 = min(m_base + l16,      TQ - 1);
    const int arow1 = min(m_base + 16 + l16, TQ - 1);
    const _Float16* aptr0 = qh + ((size_t)n * TQ + arow0) * D_DIM + half * 8;
    const _Float16* aptr1 = qh + ((size_t)n * TQ + arow1) * D_DIM + half * 8;
    const _Float16* sbase = sh + (size_t)n * TS * D_DIM;

    float best0[8], best1[8];
    int   bidx0[8], bidx1[8];
    #pragma unroll
    for (int r = 0; r < 8; ++r) {
        best0[r] = -3.0e38f; bidx0[r] = 0;
        best1[r] = -3.0e38f; bidx1[r] = 0;
    }

    const _Float16* bls0 = &Bbuf[l16 * D_DIM + half * 8];
    const _Float16* bls1 = &Bbuf[(16 + l16) * D_DIM + half * 8];

    const int nPairs = (TS + 31) / 32;   // 172 iterations of 32 support cols

    for (int tt = 0; tt < nPairs; ++tt) {
        const int ts0 = tt * 32;

        // --- cooperative stage: 32 rows x 768 halfs = 3072 b128 chunks,
        //     24 chunks per thread (blockDim == 128) ---
        #pragma unroll 6
        for (int i = 0; i < 24; ++i) {
            const int c    = threadIdx.x + 128 * i;  // 0..3071
            const int row  = c / 96;                 // 96 chunks per row
            const int off  = (c % 96) * 8;
            const int srow = min(ts0 + row, TS - 1);
            h8 d = *(const h8*)(sbase + (size_t)srow * D_DIM + off);
            *(h8*)(&Bbuf[row * D_DIM + off]) = d;
        }
        __syncthreads();   // B tile ready

        if (tt + 1 < nPairs) {   // warm L0/L2 for the next stage
            const int prow0 = min(ts0 + 32 + lane, TS - 1);
            __builtin_prefetch(sbase + (size_t)prow0 * D_DIM, 0, 1);
        }

        v8f acc00 = {0,0,0,0,0,0,0,0}, acc01 = {0,0,0,0,0,0,0,0};
        v8f acc10 = {0,0,0,0,0,0,0,0}, acc11 = {0,0,0,0,0,0,0,0};

        #pragma unroll 2
        for (int k = 0; k < D_DIM; k += 32) {
            h8 a0lo = *(const h8*)(aptr0 + k);
            h8 a0hi = *(const h8*)(aptr0 + k + 16);
            h8 a1lo = *(const h8*)(aptr1 + k);
            h8 a1hi = *(const h8*)(aptr1 + k + 16);
            h8 b0lo = *(const h8*)(bls0 + k);        // ds_load_b128
            h8 b0hi = *(const h8*)(bls0 + k + 16);
            h8 b1lo = *(const h8*)(bls1 + k);
            h8 b1hi = *(const h8*)(bls1 + k + 16);
            v16h A0, A1, B0, B1;
            #pragma unroll
            for (int i = 0; i < 8; ++i) {
                A0[i] = a0lo[i]; A0[i + 8] = a0hi[i];
                A1[i] = a1lo[i]; A1[i + 8] = a1hi[i];
                B0[i] = b0lo[i]; B0[i + 8] = b0hi[i];
                B1[i] = b1lo[i]; B1[i + 8] = b1hi[i];
            }
            acc00 = __builtin_amdgcn_wmma_f32_16x16x32_f16(false, A0, false, B0, (short)0, acc00, false, false);
            acc01 = __builtin_amdgcn_wmma_f32_16x16x32_f16(false, A0, false, B1, (short)0, acc01, false, false);
            acc10 = __builtin_amdgcn_wmma_f32_16x16x32_f16(false, A1, false, B0, (short)0, acc10, false, false);
            acc11 = __builtin_amdgcn_wmma_f32_16x16x32_f16(false, A1, false, B1, (short)0, acc11, false, false);
        }

        // C layout: lane%16 == column. Update lower-index column tile first
        // so strict '>' keeps the first occurrence on ties (argmin semantics).
        const int c0 = ts0 + l16;
        const int c1 = ts0 + 16 + l16;
        if (c0 < TS) {
            #pragma unroll
            for (int r = 0; r < 8; ++r) {
                if (acc00[r] > best0[r]) { best0[r] = acc00[r]; bidx0[r] = c0; }
                if (acc10[r] > best1[r]) { best1[r] = acc10[r]; bidx1[r] = c0; }
            }
        }
        if (c1 < TS) {
            #pragma unroll
            for (int r = 0; r < 8; ++r) {
                if (acc01[r] > best0[r]) { best0[r] = acc01[r]; bidx0[r] = c1; }
                if (acc11[r] > best1[r]) { best1[r] = acc11[r]; bidx1[r] = c1; }
            }
        }
        __syncthreads();   // everyone done reading Bbuf before next overwrite
    }

    // arg-reduce across the 16 lanes of each half (masks 1,2,4,8 stay in-half)
    #pragma unroll
    for (int off = 8; off >= 1; off >>= 1) {
        #pragma unroll
        for (int r = 0; r < 8; ++r) {
            float ob0 = __shfl_xor(best0[r], off, 32);
            int   oi0 = __shfl_xor(bidx0[r], off, 32);
            if (ob0 > best0[r] || (ob0 == best0[r] && oi0 < bidx0[r])) {
                best0[r] = ob0; bidx0[r] = oi0;
            }
            float ob1 = __shfl_xor(best1[r], off, 32);
            int   oi1 = __shfl_xor(bidx1[r], off, 32);
            if (ob1 > best1[r] || (ob1 == best1[r] && oi1 < bidx1[r])) {
                best1[r] = ob1; bidx1[r] = oi1;
            }
        }
    }

    if (l16 == 0) {  // lane 0 -> C rows 0..7 ; lane 16 -> C rows 8..15
        const float hc = hcls[n];
        const float bb = bhead[0];
        #pragma unroll
        for (int t = 0; t < 2; ++t) {
            const float* bst = t ? best1 : best0;
            const int*   bix = t ? bidx1 : bidx0;
            #pragma unroll
            for (int r = 0; r < 8; ++r) {
                const int row = m_base + 16 * t + r + 8 * half;
                if (row < TQ) {
                    const float dmin  = 1.0f - bst[r];
                    const float logit = hq[(size_t)n * TQ + row]
                                      + hs[(size_t)n * TS + bix[r]] + hc + bb;
                    const float pred  = 1.0f / (1.0f + __expf(-logit));
                    const size_t o = (size_t)n * TQ + row;
                    out0[o] = pred * dmin;
                    out1[o] = pred;
                }
            }
        }
    }
}

extern "C" void kernel_launch(void* const* d_in, const int* in_sizes, int n_in,
                              void* d_out, int out_size, void* d_ws, size_t ws_size,
                              hipStream_t stream) {
    const float* xq   = (const float*)d_in[0];   // [16,1369,768]
    const float* xs   = (const float*)d_in[1];   // [16,5476,768]
    const float* xcls = (const float*)d_in[2];   // [16,4,768]
    const float* W    = (const float*)d_in[3];   // [2304,1]
    const float* b    = (const float*)d_in[4];   // [1]
    float* out = (float*)d_out;                  // [2 * 16*1369]

    // workspace layout (bytes, 256-aligned sections)
    char* ws = (char*)d_ws;
    const size_t qh_bytes = (size_t)N_B * TQ * D_DIM * sizeof(_Float16); // 33.6 MB
    const size_t sh_bytes = (size_t)N_B * TS * D_DIM * sizeof(_Float16); // 134.6 MB
    const size_t hq_bytes = (size_t)N_B * TQ * sizeof(float);
    const size_t hs_bytes = (size_t)N_B * TS * sizeof(float);
    size_t off = 0;
    _Float16* qh  = (_Float16*)(ws + off); off += (qh_bytes + 255) & ~(size_t)255;
    _Float16* shp = (_Float16*)(ws + off); off += (sh_bytes + 255) & ~(size_t)255;
    float* hq   = (float*)(ws + off); off += (hq_bytes + 255) & ~(size_t)255;
    float* hs   = (float*)(ws + off); off += (hs_bytes + 255) & ~(size_t)255;
    float* hcls = (float*)(ws + off);

    const int q_rows = N_B * TQ;     // 21904
    const int s_rows = N_B * TS;     // 87616

    norm_rows_kernel<<<(q_rows + 7) / 8, 256, 0, stream>>>(xq, W,         qh,  hq, q_rows);
    norm_rows_kernel<<<(s_rows + 7) / 8, 256, 0, stream>>>(xs, W + D_DIM, shp, hs, s_rows);
    cls_head_kernel<<<N_B, 32, 0, stream>>>(xcls, W + 2 * D_DIM, hcls);

    dim3 grid((TQ + 127) / 128, N_B);   // (11, 16)
    sim_argmax_kernel<<<grid, 128, 0, stream>>>(qh, shp, hq, hs, hcls, b,
                                                out, out + (size_t)N_B * TQ);
}
